// MoE_53987738911322
// MI455X (gfx1250) — matrix operand order repaired and checked by reading, not compiled, not run on hardware
//
#include <hip/hip_runtime.h>
#include <hip/hip_bf16.h>

// ---------------- problem constants ----------------
#define NTOK 8192
#define CDIM 768
#define IDIM 3072
#define NEXP 8
#define CAP  1280
#define ECAP (NEXP * CAP)      // 10240
#define YELEMS ((size_t)NTOK * CDIM)

typedef __attribute__((ext_vector_type(8)))  float   v8f;
typedef __attribute__((ext_vector_type(16))) __bf16  v16bf;
typedef __attribute__((ext_vector_type(8)))  __bf16  v8bf;

static __device__ __forceinline__ __bf16 f2bf(float f) { return (__bf16)f; }

// async copy of 16 bytes per lane: global -> LDS, tracked by ASYNCcnt
static __device__ __forceinline__ void async_copy_b128(unsigned lds_byte_addr,
                                                       unsigned long long gaddr) {
  asm volatile("global_load_async_to_lds_b128 %0, %1, off"
               :: "v"(lds_byte_addr), "v"(gaddr)
               : "memory");
}
#define WAIT_ASYNC(n) asm volatile("s_wait_asynccnt " #n ::: "memory")

static __device__ __forceinline__ v16bf ld_frag(const __bf16* p) {
  v8bf lo = *(const v8bf*)p;
  v8bf hi = *(const v8bf*)(p + 8);
  return __builtin_shufflevector(lo, hi, 0, 1, 2, 3, 4, 5, 6, 7,
                                 8, 9, 10, 11, 12, 13, 14, 15);
}

// ---------------- K0: weight pack (fp32 -> bf16, transpose to N-major) ----------------
// Wg,Wu: [E][C][I] -> [E][I][C] ; Wp: [E][I][C] -> [E][C][I]
__global__ void moe_pack_weights(const float* __restrict__ Wg,
                                 const float* __restrict__ Wu,
                                 const float* __restrict__ Wp,
                                 __bf16* __restrict__ Wgt,
                                 __bf16* __restrict__ Wut,
                                 __bf16* __restrict__ Wpt,
                                 float* __restrict__ importance) {
  if (blockIdx.x == 0 && threadIdx.x < NEXP) importance[threadIdx.x] = 0.0f;
  const size_t total = (size_t)NEXP * CDIM * IDIM;
  for (size_t i = (size_t)blockIdx.x * blockDim.x + threadIdx.x; i < total;
       i += (size_t)gridDim.x * blockDim.x) {
    size_t c = i % CDIM;
    size_t n = (i / CDIM) % IDIM;
    size_t e = i / ((size_t)CDIM * IDIM);
    size_t src = e * (size_t)CDIM * IDIM + c * IDIM + n;
    Wgt[i] = f2bf(Wg[src]);
    Wut[i] = f2bf(Wu[src]);
    size_t n2 = i % IDIM;
    size_t c2 = (i / IDIM) % CDIM;
    size_t e2 = i / ((size_t)CDIM * IDIM);
    Wpt[i] = f2bf(Wp[e2 * (size_t)CDIM * IDIM + n2 * CDIM + c2]);
  }
}

// ---------------- K1: router (one wave32 per token) ----------------
__global__ void moe_router(const float* __restrict__ xf,
                           const float* __restrict__ Wr,
                           int* __restrict__ idx1, int* __restrict__ idx2,
                           float* __restrict__ gate1, float* __restrict__ gate2,
                           float* __restrict__ importance) {
  const int lane  = threadIdx.x & 31;
  const int token = blockIdx.x * 8 + (threadIdx.x >> 5);
  const float* xr = xf + (size_t)token * CDIM;

  float acc[NEXP];
#pragma unroll
  for (int e = 0; e < NEXP; ++e) acc[e] = 0.0f;
  for (int c = lane; c < CDIM; c += 32) {
    float xv = xr[c];
    const float* wr = Wr + (size_t)c * NEXP;
#pragma unroll
    for (int e = 0; e < NEXP; ++e) acc[e] += xv * wr[e];
  }
#pragma unroll
  for (int off = 16; off > 0; off >>= 1) {
#pragma unroll
    for (int e = 0; e < NEXP; ++e) acc[e] += __shfl_down(acc[e], off, 32);
  }
  if (lane == 0) {
    float mx = acc[0];
#pragma unroll
    for (int e = 1; e < NEXP; ++e) mx = fmaxf(mx, acc[e]);
    float p[NEXP], s = 0.0f;
#pragma unroll
    for (int e = 0; e < NEXP; ++e) { p[e] = __expf(acc[e] - mx); s += p[e]; }
    float inv = 1.0f / s;
#pragma unroll
    for (int e = 0; e < NEXP; ++e) p[e] *= inv;
    int i1 = 0, i2 = 0;
    float b1 = -1.0f, b2 = -1.0f;
#pragma unroll
    for (int e = 0; e < NEXP; ++e) {
      if (p[e] > b1)      { b2 = b1; i2 = i1; b1 = p[e]; i1 = e; }
      else if (p[e] > b2) { b2 = p[e]; i2 = e; }
    }
    float gs = fmaxf(b1 + b2, 1e-9f);
    idx1[token]  = i1;
    idx2[token]  = i2;
    gate1[token] = b1 / gs;
    gate2[token] = b2 / gs;
#pragma unroll
    for (int e = 0; e < NEXP; ++e) atomicAdd(&importance[e], p[e]);
  }
}

// ---------------- K2: token-order rank scan (one wave per expert) + aux loss ----------------
__global__ void moe_scan(const int* __restrict__ idx1, const int* __restrict__ idx2,
                         int* __restrict__ pos1, int* __restrict__ pos2,
                         const float* __restrict__ importance,
                         float* __restrict__ aux_out) {
  __shared__ int s_count1[NEXP];
  __shared__ int s_tot1[NEXP];
  __shared__ int s_tot2[NEXP];
  const int e    = threadIdx.x >> 5;
  const int lane = threadIdx.x & 31;
  const unsigned below = (1u << lane) - 1u;

  unsigned run = 0;
  for (int base = 0; base < NTOK; base += 32) {
    int t = base + lane;
    bool m = (idx1[t] == e);
    unsigned mask = (unsigned)__ballot(m);
    if (m) pos1[t] = (int)(run + __popc(mask & below));
    run += __popc(mask);
  }
  if (lane == 0) { s_tot1[e] = (int)run; s_count1[e] = (run < CAP) ? (int)run : CAP; }
  __syncthreads();

  unsigned run2 = 0;
  const int c1 = s_count1[e];
  for (int base = 0; base < NTOK; base += 32) {
    int t = base + lane;
    bool m = (idx2[t] == e);
    unsigned mask = (unsigned)__ballot(m);
    if (m) pos2[t] = (int)(run2 + __popc(mask & below)) + c1;
    run2 += __popc(mask);
  }
  if (lane == 0) s_tot2[e] = (int)run2;
  __syncthreads();

  if (threadIdx.x == 0) {
    float aux = 0.0f;
#pragma unroll
    for (int k = 0; k < NEXP; ++k) {
      float imp  = importance[k] / (float)NTOK;
      float load = (float)(s_tot1[k] + s_tot2[k]) / (float)(NTOK * 2);
      aux += imp * load;
    }
    aux_out[0] = aux * (float)NEXP * 0.01f;
  }
}

// ---------------- K3a: zero dispatch buffer ----------------
__global__ void moe_zero_xb(unsigned int* __restrict__ xb_u32) {
  const size_t total = (size_t)ECAP * CDIM / 2;
  for (size_t i = (size_t)blockIdx.x * blockDim.x + threadIdx.x; i < total;
       i += (size_t)gridDim.x * blockDim.x)
    xb_u32[i] = 0u;
}

// ---------------- K3b: dispatch (scatter tokens into capacity slots, fp32->bf16) ----------------
__global__ void moe_dispatch(const float* __restrict__ xf,
                             const int* __restrict__ idx1, const int* __restrict__ idx2,
                             const int* __restrict__ pos1, const int* __restrict__ pos2,
                             __bf16* __restrict__ xb) {
  const int t = blockIdx.x;
  const int p1 = pos1[t], p2 = pos2[t];
  const bool m1 = p1 < CAP, m2 = p2 < CAP;
  const size_t s1 = (size_t)(idx1[t] * CAP + p1) * CDIM;
  const size_t s2 = (size_t)(idx2[t] * CAP + p2) * CDIM;
  const float* src = xf + (size_t)t * CDIM;
  for (int c = threadIdx.x; c < CDIM; c += blockDim.x) {
    __bf16 v = f2bf(src[c]);
    if (m1) xb[s1 + c] = v;
    if (m2) xb[s2 + c] = v;
  }
}

// ---------------- K4: fused gate/up WMMA GEMM + SwiGLU ----------------
// Block = 8 waves = 128 M rows x 64 N cols; B tiles double-buffered in LDS via
// global_load_async_to_lds_b128; K staged in 64-element chunks.
__global__ void moe_gemm1(const __bf16* __restrict__ xb,
                          const __bf16* __restrict__ Wgt,
                          const __bf16* __restrict__ Wut,
                          __bf16* __restrict__ act) {
  __shared__ __bf16 sB[2 * 2 * 64 * 72];          // [buf][mat][col][72]
  const int tid  = threadIdx.x;
  const int lane = tid & 31;
  const int wv   = tid >> 5;                       // 0..7
  const int b    = blockIdx.x;
  const int nT   = b % 48;                         // 48*64 = 3072
  const int mB   = (b / 48) % 10;                  // 10*128 = 1280
  const int e    = b / 480;
  const int r15  = lane & 15;
  const int half = lane >> 4;
  const int colBase = nT * 64;

  const __bf16* wgB  = Wgt + (size_t)e * IDIM * CDIM;
  const __bf16* wuB  = Wut + (size_t)e * IDIM * CDIM;
  const __bf16* Arow = xb + (size_t)(e * CAP + (mB * 8 + wv) * 16 + r15) * CDIM;
  const unsigned ldsBase = (unsigned)(size_t)&sB[0];

  // stage s covers K = [s*64, s*64+64); per wave: 4 async b128 issues per stage
  auto issue = [&](int s, int buf) {
#pragma unroll
    for (int m = 0; m < 2; ++m) {
      const __bf16* src = m ? wuB : wgB;
#pragma unroll
      for (int j = 0; j < 2; ++j) {
        int chunk = j * 256 + tid;                 // 512 chunks of 16B
        int c = chunk >> 3, piece = chunk & 7;
        unsigned loff = ldsBase +
            (unsigned)(((buf * 2 + m) * (64 * 72) + c * 72 + piece * 8) * 2);
        unsigned long long g = (unsigned long long)(size_t)
            (src + (size_t)(colBase + c) * CDIM + (size_t)s * 64 + piece * 8);
        async_copy_b128(loff, g);
      }
    }
  };

  v8f accG[4] = {};
  v8f accU[4] = {};
  const int NS = CDIM / 64;                        // 12 stages
  issue(0, 0);
  for (int s = 0; s < NS; ++s) {
    const int buf = s & 1;
    if (s + 1 < NS) { issue(s + 1, buf ^ 1); WAIT_ASYNC(4); }
    else            { WAIT_ASYNC(0); }
    __syncthreads();
    __builtin_prefetch(Arow + s * 64 + 128, 0, 1);
#pragma unroll
    for (int q = 0; q < 2; ++q) {
      const int kk = s * 64 + q * 32;
      // A fragment: V0-7 = K[kb..kb+15], kb = half*16
      v8bf alo = *(const v8bf*)(Arow + kk + half * 16);
      v8bf ahi = *(const v8bf*)(Arow + kk + half * 16 + 8);
      v16bf a = __builtin_shufflevector(alo, ahi, 0, 1, 2, 3, 4, 5, 6, 7,
                                        8, 9, 10, 11, 12, 13, 14, 15);
      v16bf fg[4], fu[4];
#pragma unroll
      for (int t = 0; t < 4; ++t) {
        const int col = t * 16 + r15;
        fg[t] = ld_frag(&sB[(buf * 2 + 0) * (64 * 72) + col * 72 + q * 32 + half * 16]);
        fu[t] = ld_frag(&sB[(buf * 2 + 1) * (64 * 72) + col * 72 + q * 32 + half * 16]);
      }
#pragma unroll
      for (int t = 0; t < 4; ++t) {
        accG[t] = __builtin_amdgcn_wmma_f32_16x16x32_bf16(false, a, false, fg[t],
                                                          (short)0, accG[t], false, false);
        accU[t] = __builtin_amdgcn_wmma_f32_16x16x32_bf16(false, a, false, fu[t],
                                                          (short)0, accU[t], false, false);
      }
    }
    __syncthreads();
  }

  // C/D layout: VGPR v -> M = half*8 + v, N = lane&15
  const int rowb = e * CAP + (mB * 8 + wv) * 16 + half * 8;
#pragma unroll
  for (int t = 0; t < 4; ++t) {
#pragma unroll
    for (int v = 0; v < 8; ++v) {
      float g = accG[t][v];
      float u = accU[t][v];
      float sv = g / (1.0f + __expf(-g));          // silu
      act[(size_t)(rowb + v) * IDIM + nT * 64 + t * 16 + r15] = f2bf(sv * u);
    }
  }
}

// ---------------- K5: down-projection WMMA GEMM (LDS-staged B) ----------------
__global__ void moe_gemm2(const __bf16* __restrict__ act,
                          const __bf16* __restrict__ Wpt,
                          float* __restrict__ outbuf) {
  __shared__ __bf16 sB[2 * 64 * 72];               // [buf][col][72]
  const int tid  = threadIdx.x;
  const int lane = tid & 31;
  const int wv   = tid >> 5;
  const int b    = blockIdx.x;
  const int nT   = b % 12;                         // 12*64 = 768
  const int mB   = (b / 12) % 10;
  const int e    = b / 120;
  const int r15  = lane & 15;
  const int half = lane >> 4;
  const int colBase = nT * 64;

  const __bf16* wpB  = Wpt + (size_t)e * CDIM * IDIM;
  const __bf16* Arow = act + (size_t)(e * CAP + (mB * 8 + wv) * 16 + r15) * IDIM;
  const unsigned ldsBase = (unsigned)(size_t)&sB[0];

  auto issue = [&](int s, int buf) {
#pragma unroll
    for (int j = 0; j < 2; ++j) {
      int chunk = j * 256 + tid;
      int c = chunk >> 3, piece = chunk & 7;
      unsigned loff = ldsBase + (unsigned)((buf * (64 * 72) + c * 72 + piece * 8) * 2);
      unsigned long long g = (unsigned long long)(size_t)
          (wpB + (size_t)(colBase + c) * IDIM + (size_t)s * 64 + piece * 8);
      async_copy_b128(loff, g);
    }
  };

  v8f acc[4] = {};
  const int NS = IDIM / 64;                        // 48 stages
  issue(0, 0);
  for (int s = 0; s < NS; ++s) {
    const int buf = s & 1;
    if (s + 1 < NS) { issue(s + 1, buf ^ 1); WAIT_ASYNC(2); }
    else            { WAIT_ASYNC(0); }
    __syncthreads();
    __builtin_prefetch(Arow + s * 64 + 128, 0, 1);
#pragma unroll
    for (int q = 0; q < 2; ++q) {
      const int kk = s * 64 + q * 32;
      v8bf alo = *(const v8bf*)(Arow + kk + half * 16);
      v8bf ahi = *(const v8bf*)(Arow + kk + half * 16 + 8);
      v16bf a = __builtin_shufflevector(alo, ahi, 0, 1, 2, 3, 4, 5, 6, 7,
                                        8, 9, 10, 11, 12, 13, 14, 15);
      v16bf fb[4];
#pragma unroll
      for (int t = 0; t < 4; ++t) {
        const int col = t * 16 + r15;
        fb[t] = ld_frag(&sB[buf * (64 * 72) + col * 72 + q * 32 + half * 16]);
      }
#pragma unroll
      for (int t = 0; t < 4; ++t)
        acc[t] = __builtin_amdgcn_wmma_f32_16x16x32_bf16(false, a, false, fb[t],
                                                         (short)0, acc[t], false, false);
    }
    __syncthreads();
  }

  const int rowb = e * CAP + (mB * 8 + wv) * 16 + half * 8;
#pragma unroll
  for (int t = 0; t < 4; ++t) {
#pragma unroll
    for (int v = 0; v < 8; ++v)
      outbuf[(size_t)(rowb + v) * CDIM + nT * 64 + t * 16 + r15] = acc[t][v];
  }
}

// ---------------- K6: combine (gather + masked gate renormalization) ----------------
__global__ void moe_combine(const float* __restrict__ outbuf,
                            const int* __restrict__ idx1, const int* __restrict__ idx2,
                            const int* __restrict__ pos1, const int* __restrict__ pos2,
                            const float* __restrict__ gate1, const float* __restrict__ gate2,
                            float* __restrict__ y) {
  const int t = blockIdx.x;
  const int p1 = pos1[t], p2 = pos2[t];
  const bool m1 = p1 < CAP, m2 = p2 < CAP;
  float g1 = m1 ? gate1[t] : 0.0f;
  float g2 = m2 ? gate2[t] : 0.0f;
  float denom = fmaxf(g1 + g2, 1e-9f);
  g1 /= denom;
  g2 /= denom;
  int s1 = idx1[t] * CAP + p1; s1 = s1 < (ECAP - 1) ? s1 : (ECAP - 1);
  int s2 = idx2[t] * CAP + p2; s2 = s2 < (ECAP - 1) ? s2 : (ECAP - 1);
  const float* r1 = outbuf + (size_t)s1 * CDIM;
  const float* r2 = outbuf + (size_t)s2 * CDIM;
  float* yr = y + (size_t)t * CDIM;
  for (int c = threadIdx.x; c < CDIM; c += blockDim.x)
    yr[c] = g1 * r1[c] + g2 * r2[c];
}

// ---------------- host-side launch ----------------
extern "C" void kernel_launch(void* const* d_in, const int* in_sizes, int n_in,
                              void* d_out, int out_size, void* d_ws, size_t ws_size,
                              hipStream_t stream) {
  const float* x  = (const float*)d_in[0];
  const float* Wr = (const float*)d_in[1];
  const float* Wg = (const float*)d_in[2];
  const float* Wu = (const float*)d_in[3];
  const float* Wp = (const float*)d_in[4];
  float* y = (float*)d_out;

  char* w = (char*)d_ws;
  size_t off = 0;
  auto take = [&](size_t bytes) -> void* {
    void* p = w + off;
    off = (off + bytes + 255) & ~(size_t)255;
    return p;
  };
  int*    idx1  = (int*)take((size_t)NTOK * 4);
  int*    idx2  = (int*)take((size_t)NTOK * 4);
  float*  gate1 = (float*)take((size_t)NTOK * 4);
  float*  gate2 = (float*)take((size_t)NTOK * 4);
  int*    pos1  = (int*)take((size_t)NTOK * 4);
  int*    pos2  = (int*)take((size_t)NTOK * 4);
  float*  imp   = (float*)take((size_t)NEXP * 4);
  __bf16* xb    = (__bf16*)take((size_t)ECAP * CDIM * 2);
  __bf16* Wgt   = (__bf16*)take((size_t)NEXP * IDIM * CDIM * 2);
  __bf16* Wut   = (__bf16*)take((size_t)NEXP * IDIM * CDIM * 2);
  __bf16* Wpt   = (__bf16*)take((size_t)NEXP * CDIM * IDIM * 2);
  __bf16* actb  = (__bf16*)take((size_t)ECAP * IDIM * 2);
  float*  outb  = (float*)take((size_t)ECAP * CDIM * 4);

  moe_pack_weights<<<4096, 256, 0, stream>>>(Wg, Wu, Wp, Wgt, Wut, Wpt, imp);
  moe_router<<<NTOK / 8, 256, 0, stream>>>(x, Wr, idx1, idx2, gate1, gate2, imp);
  moe_scan<<<1, 256, 0, stream>>>(idx1, idx2, pos1, pos2, imp, y + YELEMS);
  moe_zero_xb<<<2048, 256, 0, stream>>>((unsigned int*)xb);
  moe_dispatch<<<NTOK, 256, 0, stream>>>(x, idx1, idx2, pos1, pos2, xb);
  moe_gemm1<<<(NEXP * 10 * 48), 256, 0, stream>>>(xb, Wgt, Wut, actb);
  moe_gemm2<<<(NEXP * 10 * 12), 256, 0, stream>>>(actb, Wpt, outb);
  moe_combine<<<NTOK, 256, 0, stream>>>(outb, idx1, idx2, pos1, pos2, gate1, gate2, y);
}
